// SA_fusion_35931696398990
// MI455X (gfx1250) — matrix operand order repaired and checked by reading, not compiled, not run on hardware
//
#include <hip/hip_runtime.h>

// ---------------------------------------------------------------------------
// SANet-style style-attention fusion for gfx1250 (MI455X), bf16 WMMA pipeline.
// B=4, C=512, H=W=64 (HW=4096).
// ---------------------------------------------------------------------------

typedef __bf16 bf16_t;
typedef __attribute__((ext_vector_type(16))) __bf16          v16bf;
typedef __attribute__((ext_vector_type(8)))  float           v8f;
typedef __attribute__((ext_vector_type(8)))  unsigned short  u16x8;
typedef __attribute__((ext_vector_type(16))) unsigned short  u16x16;

static constexpr int NB = 4;
static constexpr int CH = 512;
static constexpr int NP = 4096;   // H*W

__device__ __forceinline__ unsigned short f2bf(float f) {
  unsigned u = __builtin_bit_cast(unsigned, f);
  unsigned r = u + 0x7FFFu + ((u >> 16) & 1u);   // round-to-nearest-even
  return (unsigned short)(r >> 16);
}

__device__ __forceinline__ v8f wmma_bf16(v16bf a, v16bf b, v8f c) {
  // emits v_wmma_f32_16x16x32_bf16
  return __builtin_amdgcn_wmma_f32_16x16x32_bf16(false, a, false, b,
                                                 (short)0, c, false, false);
}

__device__ __forceinline__ v16bf mk_a(const unsigned short* p) {
  // A fragment: halves[0..7] at p, halves[8..15] at p+8 (two b128 loads)
  u16x8 a0 = *(const u16x8*)(p);
  u16x8 a1 = *(const u16x8*)(p + 8);
  u16x16 v = __builtin_shufflevector(a0, a1, 0,1,2,3,4,5,6,7,8,9,10,11,12,13,14,15);
  return __builtin_bit_cast(v16bf, v);
}

__device__ __forceinline__ v16bf mk_b(const unsigned short* p) {
  // B fragment: 16 contiguous halves (32B, two b128 loads)
  u16x16 v = *(const u16x16*)(p);
  return __builtin_bit_cast(v16bf, v);
}

// ---------------------------------------------------------------------------
// 1) Instance-norm statistics: mean + rstd per (b,c) over HW, unbiased var.
// ---------------------------------------------------------------------------
__global__ void k_stats(const float* __restrict__ x,
                        float* __restrict__ mean, float* __restrict__ rstd) {
  int row = blockIdx.x;                         // b*CH + c
  const float* p = x + (size_t)row * NP;
  float s = 0.f, ss = 0.f;
  for (int i = threadIdx.x; i < NP; i += 256) { float v = p[i]; s += v; ss += v * v; }
  __shared__ float sm[256], sm2[256];
  sm[threadIdx.x] = s; sm2[threadIdx.x] = ss; __syncthreads();
  for (int o = 128; o > 0; o >>= 1) {
    if (threadIdx.x < o) { sm[threadIdx.x] += sm[threadIdx.x + o];
                           sm2[threadIdx.x] += sm2[threadIdx.x + o]; }
    __syncthreads();
  }
  if (threadIdx.x == 0) {
    float m   = sm[0] / (float)NP;
    float var = (sm2[0] - (float)NP * m * m) / (float)(NP - 1) + 1e-5f;
    mean[row] = m;
    rstd[row] = rsqrtf(var);
  }
}

// ---------------------------------------------------------------------------
// 2) Fold instance-norm into conv weights (per batch):
//    W'[b][o][c] = bf16(W[o][c]*rstd[b][c]);  b'[b][o] = b[o] - sum_c W'*mean
// ---------------------------------------------------------------------------
__global__ void k_fold_w(const float* __restrict__ w, const float* __restrict__ bias,
                         const float* __restrict__ mean, const float* __restrict__ rstd,
                         unsigned short* __restrict__ wOut, float* __restrict__ bOut) {
  int o = blockIdx.x, b = blockIdx.y;
  const float* mrow = mean + b * CH;
  const float* rrow = rstd + b * CH;
  const float* wrow = w + (size_t)o * CH;
  unsigned short* out = wOut + ((size_t)b * CH + o) * CH;
  float acc = 0.f;
  for (int c = threadIdx.x; c < CH; c += 256) {
    float ws = wrow[c] * rrow[c];
    out[c] = f2bf(ws);
    acc += ws * mrow[c];
  }
  __shared__ float sm[256];
  sm[threadIdx.x] = acc; __syncthreads();
  for (int o2 = 128; o2 > 0; o2 >>= 1) {
    if (threadIdx.x < o2) sm[threadIdx.x] += sm[threadIdx.x + o2];
    __syncthreads();
  }
  if (threadIdx.x == 0) bOut[b * CH + o] = bias[o] - sm[0];
}

// 2b) Plain f32 -> bf16 conversion (w3, wrs).
__global__ void k_cvt(const float* __restrict__ in, unsigned short* __restrict__ out, int n) {
  int i = blockIdx.x * 256 + threadIdx.x;
  if (i < n) out[i] = f2bf(in[i]);
}

// ---------------------------------------------------------------------------
// 3) Transpose-convert x[b][c][p] f32 -> xT[b][p][c] bf16 (32x32 LDS tiles).
// ---------------------------------------------------------------------------
__global__ void k_transpose(const float* __restrict__ x, unsigned short* __restrict__ xt) {
  __shared__ unsigned short tile[32][33];
  int b = blockIdx.z;
  int p0 = blockIdx.x * 32, c0 = blockIdx.y * 32;
  const float* xb = x + (size_t)b * CH * NP;
  unsigned short* xtb = xt + (size_t)b * NP * CH;
  for (int i = threadIdx.y; i < 32; i += 8)
    tile[i][threadIdx.x] = f2bf(xb[(size_t)(c0 + i) * NP + p0 + threadIdx.x]);
  __syncthreads();
  for (int i = threadIdx.y; i < 32; i += 8)
    xtb[(size_t)(p0 + i) * CH + c0 + threadIdx.x] = tile[threadIdx.x][i];
}

// ---------------------------------------------------------------------------
// 4) q/k/v GEMM:  D[o,p] = sum_c W'[o][c] * xT[p][c] + bias[o]
//    A = W' tile (M=o,K=c, [o][c] row-major), B = xT tile (K=c,N=p, [p][c]).
//    Double-buffered K loop so loads overlap WMMA.
// ---------------------------------------------------------------------------
template <int OUT_POS_MAJOR>
__global__ void k_gemm_qkv(const unsigned short* __restrict__ W, size_t wBatchStride,
                           const float* __restrict__ bias, int biasBatchStride,
                           const unsigned short* __restrict__ xT,
                           unsigned short* __restrict__ out) {
  int b    = blockIdx.z;
  int wave = threadIdx.x >> 5;
  int lane = threadIdx.x & 31;
  int hi   = lane >> 4, lr = lane & 15;
  int p0   = blockIdx.x * 16;
  int o0   = (blockIdx.y * 8 + wave) * 16;

  const unsigned short* wrow = W + (size_t)b * wBatchStride + (size_t)(o0 + lr) * CH + 16 * hi;
  const unsigned short* xrow = xT + ((size_t)b * NP + p0 + lr) * CH + 16 * hi;
  const float* biasb = bias + (size_t)b * biasBatchStride;

  v8f acc = (v8f)0.0f;
  v16bf a0 = mk_a(wrow), b0 = mk_b(xrow);
#pragma unroll
  for (int cb = 32; cb < CH; cb += 32) {
    v16bf a1 = mk_a(wrow + cb);
    v16bf b1 = mk_b(xrow + cb);
    acc = wmma_bf16(a0, b0, acc);
    a0 = a1; b0 = b1;
  }
  acc = wmma_bf16(a0, b0, acc);

  if (OUT_POS_MAJOR) {           // out[b][p][o], o contiguous per lane
    u16x8 vals;
#pragma unroll
    for (int r = 0; r < 8; r++) vals[r] = f2bf(acc[r] + biasb[o0 + 8 * hi + r]);
    *(u16x8*)(out + ((size_t)b * NP + p0 + lr) * CH + o0 + 8 * hi) = vals;
  } else {                       // out[b][c][p]
#pragma unroll
    for (int r = 0; r < 8; r++) {
      int o = o0 + 8 * hi + r;
      out[((size_t)b * CH + o) * NP + p0 + lr] = f2bf(acc[r] + biasb[o]);
    }
  }
}

// ---------------------------------------------------------------------------
// 5) Softmax statistics: m[b][q] = max_k S, linv[b][q] = 1/sum_k exp(S-m).
//    Q tile (16x512 bf16, 16 KB) staged in LDS once per block; A fragments
//    come from LDS (ds_load_b128), B fragments from global, double-buffered.
//    The K-step order is rotated by kt so the LDS A loads cannot be hoisted
//    (LICM'd) out of the kt loop into spilled registers.
// ---------------------------------------------------------------------------
__global__ void k_softmax_stats(const unsigned short* __restrict__ xq,
                                const unsigned short* __restrict__ xk,
                                float* __restrict__ mOut, float* __restrict__ linvOut) {
  __shared__ unsigned short Qs[16][CH];          // 16 KB
  int b = blockIdx.y;
  int q0 = blockIdx.x * 16;
  int wave = threadIdx.x >> 5, lane = threadIdx.x & 31;
  int hi = lane >> 4, lr = lane & 15;

  // cooperative stage of the Q tile (contiguous 16 KB)
  {
    const u16x8* src = (const u16x8*)(xq + ((size_t)b * NP + q0) * CH);
    u16x8* dst = (u16x8*)&Qs[0][0];
    for (int i = threadIdx.x; i < 16 * CH / 8; i += 256) dst[i] = src[i];
  }
  __syncthreads();

  const unsigned short* kbase = xk + (size_t)b * NP * CH;
  const unsigned short* arow  = &Qs[lr][16 * hi];

  float m_run[8], s_run[8];
#pragma unroll
  for (int r = 0; r < 8; r++) { m_run[r] = -3.0e38f; s_run[r] = 0.f; }

  for (int kt = 0; kt < 32; kt++) {
    int kpos = kt * 128 + wave * 16;             // waves interleave k tiles
    const unsigned short* krow = kbase + (size_t)(kpos + lr) * CH + 16 * hi;
    int rot = kt & 15;                           // anti-LICM rotation
    v8f acc = (v8f)0.0f;
    v16bf a0 = mk_a(arow + rot * 32);
    v16bf b0 = mk_b(krow + rot * 32);
#pragma unroll
    for (int i = 1; i < 16; i++) {
      int cc = (rot + i) & 15;
      v16bf a1 = mk_a(arow + cc * 32);
      v16bf b1 = mk_b(krow + cc * 32);
      acc = wmma_bf16(a0, b0, acc);
      a0 = a1; b0 = b1;
    }
    acc = wmma_bf16(a0, b0, acc);

#pragma unroll
    for (int r = 0; r < 8; r++) {
      float v = acc[r];
      float tm = v;
      for (int d = 8; d > 0; d >>= 1) tm = fmaxf(tm, __shfl_xor(tm, d, 16));
      float nm = fmaxf(m_run[r], tm);
      float e  = __expf(v - nm);
      float ts = e;
      for (int d = 8; d > 0; d >>= 1) ts += __shfl_xor(ts, d, 16);
      s_run[r] = s_run[r] * __expf(m_run[r] - nm) + ts;
      m_run[r] = nm;
    }
  }
  __shared__ float ms[8][16], ss[8][16];
  if (lr == 0) {
#pragma unroll
    for (int r = 0; r < 8; r++) { ms[wave][8 * hi + r] = m_run[r];
                                  ss[wave][8 * hi + r] = s_run[r]; }
  }
  __syncthreads();
  if (threadIdx.x < 16) {
    int q = threadIdx.x;
    float m = ms[0][q];
    for (int w = 1; w < 8; w++) m = fmaxf(m, ms[w][q]);
    float s = 0.f;
    for (int w = 0; w < 8; w++) s += ss[w][q] * __expf(ms[w][q] - m);
    mOut[b * NP + q0 + q]    = m;
    linvOut[b * NP + q0 + q] = 1.f / s;
  }
}

// ---------------------------------------------------------------------------
// 6) Attention aggregation: frs[b][p][c] = sum_k v[b][c][k] * P[q=p][k], with
//    P = exp(S - m) * linv.  Q tile staged in LDS; P tiles staged through LDS
//    (16 x 256 bf16); each wave owns 64 output channels.  Scores K-steps are
//    rotated per (it,half) to defeat LICM of the LDS A loads.
// ---------------------------------------------------------------------------
__global__ void k_attn_agg(const unsigned short* __restrict__ xq,
                           const unsigned short* __restrict__ xk,
                           const unsigned short* __restrict__ xv,
                           const float* __restrict__ mIn,
                           const float* __restrict__ linvIn,
                           unsigned short* __restrict__ frs) {
  __shared__ unsigned short Qs[16][CH];          // 16 KB
  __shared__ unsigned short P[16][256];          //  8 KB
  int b = blockIdx.y, q0 = blockIdx.x * 16;
  int wave = threadIdx.x >> 5, lane = threadIdx.x & 31;
  int hi = lane >> 4, lr = lane & 15;

  {
    const u16x8* src = (const u16x8*)(xq + ((size_t)b * NP + q0) * CH);
    u16x8* dst = (u16x8*)&Qs[0][0];
    for (int i = threadIdx.x; i < 16 * CH / 8; i += 256) dst[i] = src[i];
  }

  const unsigned short* kbase = xk + (size_t)b * NP * CH;
  const unsigned short* vbase = xv + (size_t)b * CH * NP;
  const unsigned short* arow  = &Qs[lr][16 * hi];

  float mrow[8], lrow[8];
#pragma unroll
  for (int r = 0; r < 8; r++) {
    mrow[r] = mIn[b * NP + q0 + 8 * hi + r];
    lrow[r] = linvIn[b * NP + q0 + 8 * hi + r];
  }
  int c0 = wave * 64;
  v8f acc[4];
#pragma unroll
  for (int t = 0; t < 4; t++) acc[t] = (v8f)0.0f;

  __syncthreads();                               // Qs ready

  for (int it = 0; it < 16; it++) {
    int kg0 = it * 256;
    // --- scores: each wave computes its 16x32 slice of P -> LDS ---
#pragma unroll
    for (int half = 0; half < 2; half++) {
      int kpos = kg0 + wave * 32 + half * 16;
      const unsigned short* krow = kbase + (size_t)(kpos + lr) * CH + 16 * hi;
      int rot = ((it << 1) | half) & 15;         // anti-LICM rotation
      v8f s = (v8f)0.0f;
      v16bf b0 = mk_b(krow + rot * 32);
      v16bf a0 = mk_a(arow + rot * 32);
#pragma unroll
      for (int i = 1; i < 16; i++) {
        int cc = (rot + i) & 15;
        v16bf a1 = mk_a(arow + cc * 32);
        v16bf b1 = mk_b(krow + cc * 32);
        s = wmma_bf16(a0, b0, s);
        a0 = a1; b0 = b1;
      }
      s = wmma_bf16(a0, b0, s);

      int klocal = wave * 32 + half * 16 + lr;
#pragma unroll
      for (int r = 0; r < 8; r++)
        P[8 * hi + r][klocal] = f2bf(__expf(s[r] - mrow[r]) * lrow[r]);
    }
    __syncthreads();
    // --- aggregation: acc[c,q] += V[c,k] * P^T[k,q] over this 256-k slab ---
#pragma unroll
    for (int ks = 0; ks < 8; ks++) {
      v16bf bv = mk_b(&P[lr][ks * 32 + 16 * hi]);      // ds_load_b128 x2
      int kg = kg0 + ks * 32;
#pragma unroll
      for (int t = 0; t < 4; t++) {
        const unsigned short* vrow =
            vbase + (size_t)(c0 + t * 16 + lr) * NP + kg + 16 * hi;
        acc[t] = wmma_bf16(mk_a(vrow), bv, acc[t]);
      }
    }
    __syncthreads();
  }
  // store frs[b][p][c], c contiguous per lane -> single 16B store per tile
#pragma unroll
  for (int t = 0; t < 4; t++) {
    u16x8 vals;
#pragma unroll
    for (int r = 0; r < 8; r++) vals[r] = f2bf(acc[t][r]);
    *(u16x8*)(frs + ((size_t)b * NP + q0 + lr) * CH + c0 + t * 16 + 8 * hi) = vals;
  }
}

// ---------------------------------------------------------------------------
// 7) Final conv + residual: out[b][o][p] = sum_c Wrs[o][c]*frs[b][p][c]
//                                          + brs[o] + x_fcc[b][o][p]
// ---------------------------------------------------------------------------
__global__ void k_gemm_rs(const unsigned short* __restrict__ Wrs,
                          const float* __restrict__ brs,
                          const unsigned short* __restrict__ frs,
                          const float* __restrict__ xfcc,
                          float* __restrict__ out) {
  int b = blockIdx.z;
  int wave = threadIdx.x >> 5, lane = threadIdx.x & 31;
  int hi = lane >> 4, lr = lane & 15;
  int p0 = blockIdx.x * 16;
  int o0 = (blockIdx.y * 8 + wave) * 16;
  const unsigned short* wrow = Wrs + (size_t)(o0 + lr) * CH + 16 * hi;
  const unsigned short* frow = frs + ((size_t)b * NP + p0 + lr) * CH + 16 * hi;
  v8f acc = (v8f)0.0f;
  v16bf a0 = mk_a(wrow), b0 = mk_b(frow);
#pragma unroll
  for (int cb = 32; cb < CH; cb += 32) {
    v16bf a1 = mk_a(wrow + cb);
    v16bf b1 = mk_b(frow + cb);
    acc = wmma_bf16(a0, b0, acc);
    a0 = a1; b0 = b1;
  }
  acc = wmma_bf16(a0, b0, acc);
#pragma unroll
  for (int r = 0; r < 8; r++) {
    int o = o0 + 8 * hi + r;
    size_t idx = ((size_t)b * CH + o) * NP + p0 + lr;
    out[idx] = acc[r] + brs[o] + xfcc[idx];
  }
}

// ---------------------------------------------------------------------------
// Host-side launcher
// ---------------------------------------------------------------------------
static inline size_t alignUp(size_t x) { return (x + 255) & ~(size_t)255; }

extern "C" void kernel_launch(void* const* d_in, const int* in_sizes, int n_in,
                              void* d_out, int out_size, void* d_ws, size_t ws_size,
                              hipStream_t stream) {
  (void)in_sizes; (void)n_in; (void)out_size; (void)ws_size;
  const float* x_fcc = (const float*)d_in[0];
  const float* x_fss = (const float*)d_in[1];
  const float* w1 = (const float*)d_in[2]; const float* b1 = (const float*)d_in[3];
  const float* w2 = (const float*)d_in[4]; const float* b2 = (const float*)d_in[5];
  const float* w3 = (const float*)d_in[6]; const float* b3 = (const float*)d_in[7];
  const float* wrs = (const float*)d_in[8]; const float* brs = (const float*)d_in[9];
  float* out = (float*)d_out;

  // workspace carve
  char* ws = (char*)d_ws;
  size_t off = 0;
  auto carve = [&](size_t bytes) { char* p = ws + off; off = alignUp(off + bytes); return p; };
  float* mean_c = (float*)carve((size_t)NB * CH * 4);
  float* rstd_c = (float*)carve((size_t)NB * CH * 4);
  float* mean_s = (float*)carve((size_t)NB * CH * 4);
  float* rstd_s = (float*)carve((size_t)NB * CH * 4);
  float* bias1p = (float*)carve((size_t)NB * CH * 4);
  float* bias2p = (float*)carve((size_t)NB * CH * 4);
  unsigned short* Wq = (unsigned short*)carve((size_t)NB * CH * CH * 2);
  unsigned short* Wk = (unsigned short*)carve((size_t)NB * CH * CH * 2);
  unsigned short* Wv = (unsigned short*)carve((size_t)CH * CH * 2);
  unsigned short* Wr = (unsigned short*)carve((size_t)CH * CH * 2);
  unsigned short* xcT = (unsigned short*)carve((size_t)NB * NP * CH * 2);
  unsigned short* xsT = (unsigned short*)carve((size_t)NB * NP * CH * 2);
  unsigned short* xq = (unsigned short*)carve((size_t)NB * NP * CH * 2);
  unsigned short* xk = (unsigned short*)carve((size_t)NB * NP * CH * 2);
  unsigned short* xv = (unsigned short*)carve((size_t)NB * CH * NP * 2);
  float* msm  = (float*)carve((size_t)NB * NP * 4);
  float* linv = (float*)carve((size_t)NB * NP * 4);
  unsigned short* frs = (unsigned short*)carve((size_t)NB * NP * CH * 2);

  // 1) instance-norm stats
  k_stats<<<NB * CH, 256, 0, stream>>>(x_fcc, mean_c, rstd_c);
  k_stats<<<NB * CH, 256, 0, stream>>>(x_fss, mean_s, rstd_s);
  // 2) fold norm into weights; convert plain weights
  k_fold_w<<<dim3(CH, NB), 256, 0, stream>>>(w1, b1, mean_c, rstd_c, Wq, bias1p);
  k_fold_w<<<dim3(CH, NB), 256, 0, stream>>>(w2, b2, mean_s, rstd_s, Wk, bias2p);
  k_cvt<<<(CH * CH + 255) / 256, 256, 0, stream>>>(w3, Wv, CH * CH);
  k_cvt<<<(CH * CH + 255) / 256, 256, 0, stream>>>(wrs, Wr, CH * CH);
  // 3) transpose-convert inputs to [b][p][c] bf16
  k_transpose<<<dim3(NP / 32, CH / 32, NB), dim3(32, 8), 0, stream>>>(x_fcc, xcT);
  k_transpose<<<dim3(NP / 32, CH / 32, NB), dim3(32, 8), 0, stream>>>(x_fss, xsT);
  // 4) q/k/v GEMMs (WMMA bf16)
  dim3 gg(NP / 16, CH / 128, NB);
  k_gemm_qkv<1><<<gg, 256, 0, stream>>>(Wq, (size_t)CH * CH, bias1p, CH, xcT, xq);
  k_gemm_qkv<1><<<gg, 256, 0, stream>>>(Wk, (size_t)CH * CH, bias2p, CH, xsT, xk);
  k_gemm_qkv<0><<<gg, 256, 0, stream>>>(Wv, 0, b3, 0, xsT, xv);
  // 5) softmax stats (flash-style, no score materialization)
  k_softmax_stats<<<dim3(NP / 16, NB), 256, 0, stream>>>(xq, xk, msm, linv);
  // 6) attention aggregation
  k_attn_agg<<<dim3(NP / 16, NB), 256, 0, stream>>>(xq, xk, xv, msm, linv, frs);
  // 7) final conv + bias + residual
  k_gemm_rs<<<gg, 256, 0, stream>>>(Wr, brs, frs, x_fcc, out);
}
